// DIENItemSequenceInteractionBlock_13134009991514
// MI455X (gfx1250) — matrix operand order repaired and checked by reading, not compiled, not run on hardware
//
#include <hip/hip_runtime.h>
#include <hip/hip_bf16.h>
#include <stddef.h>
#include <stdint.h>

// ---------------------------------------------------------------------------
// DIEN item-sequence interaction block for MI455X (gfx1250), wave32 + WMMA +
// async global->LDS DMA.  B=2048, T=200, E=H=128, D1=80, D2=40.
// ---------------------------------------------------------------------------

typedef __bf16 bf16_t;
typedef __attribute__((ext_vector_type(16))) __bf16 v16bf;
typedef __attribute__((ext_vector_type(8)))  float  v8f;

#define BB 2048
#define TT 200
#define EE 128
#define HH 128

union ABFrag { v16bf v; uint4 q[2]; };

__device__ __forceinline__ v8f wmma_bf16(v16bf a, v16bf b, v8f c) {
  // D = A(16x32 bf16) * B(32x16 bf16) + C(16x16 f32)
  return __builtin_amdgcn_wmma_f32_16x16x32_bf16(
      /*neg_a=*/false, a, /*neg_b=*/false, b,
      /*c_mod=*/(short)0, c, /*reuse_a=*/false, /*reuse_b=*/false);
}

// Async DMA: 16 bytes global -> LDS, tracked by ASYNCcnt (no VGPR staging).
__device__ __forceinline__ void async_g2lds_b128(void* lds_ptr, const void* gptr) {
  uint32_t loff = (uint32_t)(size_t)lds_ptr;   // generic LDS ptr low 32b = LDS addr
  asm volatile("global_load_async_to_lds_b128 %0, %1, off"
               :: "v"(loff), "v"(gptr)
               : "memory");
}
__device__ __forceinline__ void wait_async0() {
  asm volatile("s_wait_asynccnt 0x0" ::: "memory");
}

// A fragment (16x32 bf16) from a row-major LDS tile [16][ldk].
__device__ __forceinline__ v16bf lds_load_a(const bf16_t* base, int ldk, int k0) {
  const int lane = threadIdx.x & 31;
  const int m    = lane & 15;
  const int kb   = (lane & 16) ? 8 : 0;
  const bf16_t* p = base + m * ldk + k0 + kb;
  ABFrag f;
  f.q[0] = *(const uint4*)(p);
  f.q[1] = *(const uint4*)(p + 16);
  return f.v;
}

// B fragment (32x16 bf16) from an N-major (transposed) LDS weight tile [N][ldk].
__device__ __forceinline__ v16bf lds_load_b(const bf16_t* baseT, int ldk, int n0, int k0) {
  const int lane = threadIdx.x & 31;
  const int n    = n0 + (lane & 15);
  const int kb   = (lane & 16) ? 8 : 0;
  const bf16_t* p = baseT + n * ldk + k0 + kb;
  ABFrag f;
  f.q[0] = *(const uint4*)(p);
  f.q[1] = *(const uint4*)(p + 16);
  return f.v;
}

__device__ __forceinline__ float sigm(float x)  { return 1.0f / (1.0f + __expf(-x)); }
__device__ __forceinline__ float tanh_f(float x){ return 2.0f / (1.0f + __expf(-2.0f * x)) - 1.0f; }

// Stage one [16][128] fp32 timestep tile into LDS via async DMA (2 x b128/thread).
__device__ __forceinline__ void stage_x_async(float* xs, const float* __restrict__ src,
                                              int b0, int t) {
  const int tid = threadIdx.x;
#pragma unroll
  for (int e = tid * 4; e < 2048; e += 1024) {
    int r = e >> 7, c = e & 127;
    async_g2lds_b128(xs + e, src + ((size_t)(b0 + r) * TT + t) * 128 + c);
  }
}

// ---------------------------------------------------------------------------
// Prep kernels: fp32 row-major [K][N] -> bf16 N-major [N][K] in workspace.
// Run once; makes every per-block LDS weight fill a raw async b128 copy.
// ---------------------------------------------------------------------------
__global__ void conv_T_kernel(const float* __restrict__ src, bf16_t* __restrict__ dst,
                              int K, int N) {
  int i = blockIdx.x * blockDim.x + threadIdx.x;
  if (i < N * K) {
    int n = i / K, k = i - n * K;
    dst[i] = (bf16_t)src[k * N + n];
  }
}
__global__ void conv_T_pad_kernel(const float* __restrict__ src, bf16_t* __restrict__ dst,
                                  int K, int N, int Kp, int Np) {
  int i = blockIdx.x * blockDim.x + threadIdx.x;
  if (i < Np * Kp) {
    int n = i / Kp, k = i - n * Kp;
    dst[i] = (bf16_t)((n < N && k < K) ? src[k * N + n] : 0.0f);
  }
}

// ---------------------------------------------------------------------------
// Kernel 1: fused GRU.  16 batch rows/block, 8 waves x 16 H-cols, W/U in LDS,
// double-buffered async x staging.
// ---------------------------------------------------------------------------
__global__ void gru_kernel(const float* __restrict__ xseq,  // [B,T,E] fp32
                           const bf16_t* __restrict__ WTg,  // [384][128] bf16 N-major
                           const bf16_t* __restrict__ UTg,  // [384][128]
                           const float* __restrict__ bx,
                           const float* __restrict__ bh,
                           float* __restrict__ feats)       // [B,T,H]
{
  extern __shared__ __align__(16) char smem_raw[];
  bf16_t* WT  = (bf16_t*)smem_raw;          // [384][128]
  bf16_t* UT  = WT + 384 * 128;             // [384][128]
  bf16_t* xb  = UT + 384 * 128;             // [16][128] bf16
  bf16_t* hb  = xb + 16 * 128;              // [16][128] bf16
  float*  ab  = (float*)(hb + 16 * 128);    // [16] (unused here; layout shared)
  float*  xs0 = ab + 16;                    // [16][128] fp32 stage 0
  float*  xs1 = xs0 + 2048;                 // [16][128] fp32 stage 1
  float*  xsb[2] = { xs0, xs1 };

  const int tid  = threadIdx.x;
  const int wave = tid >> 5;
  const int lane = tid & 31;
  const int b0   = blockIdx.x * 16;

  // Async DMA weight fill (bf16, already transposed): 48 x b128 per thread.
  for (int i = tid * 8; i < 384 * 128; i += 2048) {
    async_g2lds_b128(WT + i, WTg + i);
    async_g2lds_b128(UT + i, UTg + i);
  }
  for (int i = tid; i < 16 * 128; i += 256) hb[i] = (bf16_t)0.0f;
  stage_x_async(xsb[0], xseq, b0, 0);

  const int nl   = lane & 15;
  const int moff = (lane & 16) ? 8 : 0;
  const int cz   = wave * 16 + nl;
  const float bz  = bx[cz]       + bh[cz];
  const float br  = bx[128 + cz] + bh[128 + cz];
  const float bxc = bx[256 + cz];
  const float bhc = bh[256 + cz];

  v8f h = {};
  wait_async0();
  __syncthreads();

  for (int t = 0; t < TT; ++t) {
    // Convert staged fp32 tile -> bf16 xb (one 8-element pass per thread).
    {
      const float* cur = xsb[t & 1];
      const int e0 = tid * 8;
#pragma unroll
      for (int j = 0; j < 8; ++j) xb[e0 + j] = (bf16_t)cur[e0 + j];
    }
    if (t + 1 < TT) stage_x_async(xsb[(t + 1) & 1], xseq, b0, t + 1);  // overlap
    __syncthreads();

    v8f az = {}, ar = {}, axc = {}, ahc = {};
#pragma unroll
    for (int kk = 0; kk < 4; ++kk) {
      const int k0 = kk * 32;
      v16bf ax = lds_load_a(xb, 128, k0);
      v16bf ah = lds_load_a(hb, 128, k0);
      az  = wmma_bf16(ax, lds_load_b(WT, 128,       wave * 16, k0), az);
      az  = wmma_bf16(ah, lds_load_b(UT, 128,       wave * 16, k0), az);
      ar  = wmma_bf16(ax, lds_load_b(WT, 128, 128 + wave * 16, k0), ar);
      ar  = wmma_bf16(ah, lds_load_b(UT, 128, 128 + wave * 16, k0), ar);
      axc = wmma_bf16(ax, lds_load_b(WT, 128, 256 + wave * 16, k0), axc);
      ahc = wmma_bf16(ah, lds_load_b(UT, 128, 256 + wave * 16, k0), ahc);
    }
    __syncthreads();

#pragma unroll
    for (int p = 0; p < 8; ++p) {
      float z  = sigm(az[p] + bz);
      float r  = sigm(ar[p] + br);
      float c  = tanh_f(axc[p] + bxc + r * (ahc[p] + bhc));
      float hn = z * h[p] + (1.0f - z) * c;
      h[p] = hn;
      int m = p + moff;
      hb[m * 128 + cz] = (bf16_t)hn;
      feats[((size_t)(b0 + m) * TT + t) * HH + cz] = hn;
    }
    wait_async0();       // our t+1 DMA done
    __syncthreads();     // everyone's DMA + hb visible
  }
}

// ---------------------------------------------------------------------------
// Kernel 2: q = PReLU(item @ att_Wi + bi).
// ---------------------------------------------------------------------------
__global__ void qproj_kernel(const float* __restrict__ item,    // [B,E]
                             const bf16_t* __restrict__ WiTg,   // [128][128] bf16 N-major
                             const float* __restrict__ bi,
                             const float* __restrict__ alpha,
                             float* __restrict__ q)             // [B,H]
{
  extern __shared__ __align__(16) char smem_raw[];
  bf16_t* WT = (bf16_t*)smem_raw;   // [128][128]
  bf16_t* xb = WT + 128 * 128;      // [16][128]

  const int tid  = threadIdx.x;
  const int wave = tid >> 5;
  const int lane = tid & 31;
  const int b0   = blockIdx.x * 16;

  for (int i = tid * 8; i < 128 * 128; i += 2048)
    async_g2lds_b128(WT + i, WiTg + i);
  for (int i = tid; i < 16 * 128; i += 256) {
    int r = i >> 7, c = i & 127;
    xb[i] = (bf16_t)item[(size_t)(b0 + r) * EE + c];
  }
  wait_async0();
  __syncthreads();

  v8f acc = {};
#pragma unroll
  for (int kk = 0; kk < 4; ++kk) {
    v16bf a = lds_load_a(xb, 128, kk * 32);
    acc = wmma_bf16(a, lds_load_b(WT, 128, wave * 16, kk * 32), acc);
  }

  const int nl   = lane & 15;
  const int moff = (lane & 16) ? 8 : 0;
  const int col  = wave * 16 + nl;
  const float bv = bi[col];
  const float al = alpha[col];
#pragma unroll
  for (int p = 0; p < 8; ++p) {
    float v = acc[p] + bv;
    v = (v >= 0.0f) ? v : al * v;
    q[(size_t)(b0 + p + moff) * HH + col] = v;
  }
}

// ---------------------------------------------------------------------------
// Kernel 3: attention MLP.  128 (b,t) rows/block, one 16-row M-tile per wave.
// ---------------------------------------------------------------------------
__global__ void attmlp_kernel(const float* __restrict__ q,       // [B,H]
                              const float* __restrict__ feats,   // [B,T,H]
                              const bf16_t* __restrict__ W1Tg,   // [80][512] bf16
                              const float* __restrict__ b1,
                              const bf16_t* __restrict__ W2Tg,   // [48][96] bf16 padded
                              const float* __restrict__ b2,
                              const float* __restrict__ W3, const float* __restrict__ b3,
                              float* __restrict__ scores)        // [B*T]
{
  extern __shared__ __align__(16) char smem_raw[];
  bf16_t* W1T  = (bf16_t*)smem_raw;       // [80][512]
  bf16_t* W2T  = W1T + 80 * 512;          // [48][96]
  bf16_t* comb = W2T + 48 * 96;           // [128][512]
  bf16_t* h1   = comb + 128 * 512;        // [128][96]
  float*  w3   = (float*)(h1 + 128 * 96); // [48]

  const int tid  = threadIdx.x;
  const int wave = tid >> 5;
  const int lane = tid & 31;
  const int row0 = blockIdx.x * 128;

  for (int i = tid * 8; i < 80 * 512; i += 2048) async_g2lds_b128(W1T + i, W1Tg + i);
  for (int i = tid * 8; i < 48 * 96;  i += 2048) async_g2lds_b128(W2T + i, W2Tg + i);
  if (tid < 48) w3[tid] = (tid < 40) ? W3[tid] : 0.0f;
  for (int i = tid; i < 128 * 96; i += 256) h1[i] = (bf16_t)0.0f;

  // comb = [q | f | q-f | q*f]
  for (int i = tid; i < 128 * 128; i += 256) {
    int r = i >> 7, c = i & 127;
    int gr = row0 + r;
    int b  = gr / TT;
    int t  = gr - b * TT;
    float qv = q[(size_t)b * HH + c];
    float fv = feats[((size_t)b * TT + t) * HH + c];
    bf16_t* cr = comb + (size_t)r * 512;
    cr[c]       = (bf16_t)qv;
    cr[128 + c] = (bf16_t)fv;
    cr[256 + c] = (bf16_t)(qv - fv);
    cr[384 + c] = (bf16_t)(qv * fv);
  }
  wait_async0();
  __syncthreads();

  const int nl   = lane & 15;
  const int moff = (lane & 16) ? 8 : 0;

  // Layer 1: [16,512] x [512,80] per wave (5 N-tiles, 16 K-steps).
  const bf16_t* arow = comb + (size_t)wave * 16 * 512;
  v8f a1[5] = {};
  for (int kk = 0; kk < 16; ++kk) {
    v16bf a = lds_load_a(arow, 512, kk * 32);
#pragma unroll
    for (int j = 0; j < 5; ++j)
      a1[j] = wmma_bf16(a, lds_load_b(W1T, 512, j * 16, kk * 32), a1[j]);
  }
#pragma unroll
  for (int j = 0; j < 5; ++j) {
    int colb = j * 16 + nl;
    float bv = b1[colb];
#pragma unroll
    for (int p = 0; p < 8; ++p) {
      float v = sigm(a1[j][p] + bv);
      h1[(wave * 16 + p + moff) * 96 + colb] = (bf16_t)v;
    }
  }
  __syncthreads();

  // Layer 2: [16,96] x [96,48] per wave.
  const bf16_t* a2row = h1 + (size_t)wave * 16 * 96;
  v8f a2[3] = {};
#pragma unroll
  for (int kk = 0; kk < 3; ++kk) {
    v16bf a = lds_load_a(a2row, 96, kk * 32);
#pragma unroll
    for (int j = 0; j < 3; ++j)
      a2[j] = wmma_bf16(a, lds_load_b(W2T, 96, j * 16, kk * 32), a2[j]);
  }

  // Layer 3: score = sigmoid(h2) . W3 + b3, half-wave shuffle reduction.
  float s[8];
#pragma unroll
  for (int p = 0; p < 8; ++p) s[p] = 0.0f;
#pragma unroll
  for (int j = 0; j < 3; ++j) {
    int colb = j * 16 + nl;
    float wv  = w3[colb];
    float b2v = (colb < 40) ? b2[colb] : 0.0f;
#pragma unroll
    for (int p = 0; p < 8; ++p) {
      float hv = (colb < 40) ? sigm(a2[j][p] + b2v) : 0.0f;
      s[p] += hv * wv;
    }
  }
#pragma unroll
  for (int off = 1; off < 16; off <<= 1) {
#pragma unroll
    for (int p = 0; p < 8; ++p) s[p] += __shfl_xor(s[p], off, 32);
  }
  if (nl == 0) {
    float bb3 = b3[0];
#pragma unroll
    for (int p = 0; p < 8; ++p) {
      int gr = row0 + wave * 16 + p + moff;
      scores[gr] = s[p] + bb3;
    }
  }
}

// ---------------------------------------------------------------------------
// Kernel 4: masked softmax over T.
// ---------------------------------------------------------------------------
__global__ void softmax_kernel(const float* __restrict__ scores,
                               const int* __restrict__ mask,
                               float* __restrict__ att) {
  __shared__ float red[256];
  const int b   = blockIdx.x;
  const int tid = threadIdx.x;
  const float FMIN = -3.402823466e38f;

  float my = FMIN;
  if (tid < TT) my = (mask[(size_t)b * TT + tid] == 1) ? scores[(size_t)b * TT + tid] : FMIN;
  red[tid] = my;
  __syncthreads();
  for (int s = 128; s > 0; s >>= 1) {
    if (tid < s) red[tid] = fmaxf(red[tid], red[tid + s]);
    __syncthreads();
  }
  float mx = red[0];
  __syncthreads();
  float e = (tid < TT) ? __expf(my - mx) : 0.0f;
  red[tid] = e;
  __syncthreads();
  for (int s = 128; s > 0; s >>= 1) {
    if (tid < s) red[tid] += red[tid + s];
    __syncthreads();
  }
  float sum = red[0];
  if (tid < TT) att[(size_t)b * TT + tid] = e / sum;
}

// ---------------------------------------------------------------------------
// Kernel 5: fused AUGRU (attention-gated update) -> final h [B,H].
// ---------------------------------------------------------------------------
__global__ void augru_kernel(const float* __restrict__ xseq,   // feats [B,T,H]
                             const float* __restrict__ att,    // [B,T]
                             const bf16_t* __restrict__ WTg,   // [384][128] bf16
                             const bf16_t* __restrict__ UTg,
                             const float* __restrict__ bx,
                             const float* __restrict__ bh,
                             float* __restrict__ out)          // [B,H]
{
  extern __shared__ __align__(16) char smem_raw[];
  bf16_t* WT  = (bf16_t*)smem_raw;
  bf16_t* UT  = WT + 384 * 128;
  bf16_t* xb  = UT + 384 * 128;
  bf16_t* hb  = xb + 16 * 128;
  float*  ab  = (float*)(hb + 16 * 128);    // [16] attention this step
  float*  xs0 = ab + 16;
  float*  xs1 = xs0 + 2048;
  float*  xsb[2] = { xs0, xs1 };

  const int tid  = threadIdx.x;
  const int wave = tid >> 5;
  const int lane = tid & 31;
  const int b0   = blockIdx.x * 16;

  for (int i = tid * 8; i < 384 * 128; i += 2048) {
    async_g2lds_b128(WT + i, WTg + i);
    async_g2lds_b128(UT + i, UTg + i);
  }
  for (int i = tid; i < 16 * 128; i += 256) hb[i] = (bf16_t)0.0f;
  stage_x_async(xsb[0], xseq, b0, 0);

  const int nl   = lane & 15;
  const int moff = (lane & 16) ? 8 : 0;
  const int cz   = wave * 16 + nl;
  const float bz  = bx[cz]       + bh[cz];
  const float br  = bx[128 + cz] + bh[128 + cz];
  const float bxc = bx[256 + cz];
  const float bhc = bh[256 + cz];

  v8f h = {};
  wait_async0();
  __syncthreads();

  for (int t = 0; t < TT; ++t) {
    {
      const float* cur = xsb[t & 1];
      const int e0 = tid * 8;
#pragma unroll
      for (int j = 0; j < 8; ++j) xb[e0 + j] = (bf16_t)cur[e0 + j];
    }
    if (tid < 16) ab[tid] = att[(size_t)(b0 + tid) * TT + t];
    if (t + 1 < TT) stage_x_async(xsb[(t + 1) & 1], xseq, b0, t + 1);
    __syncthreads();

    v8f az = {}, ar = {}, axc = {}, ahc = {};
#pragma unroll
    for (int kk = 0; kk < 4; ++kk) {
      const int k0 = kk * 32;
      v16bf ax = lds_load_a(xb, 128, k0);
      v16bf ah = lds_load_a(hb, 128, k0);
      az  = wmma_bf16(ax, lds_load_b(WT, 128,       wave * 16, k0), az);
      az  = wmma_bf16(ah, lds_load_b(UT, 128,       wave * 16, k0), az);
      ar  = wmma_bf16(ax, lds_load_b(WT, 128, 128 + wave * 16, k0), ar);
      ar  = wmma_bf16(ah, lds_load_b(UT, 128, 128 + wave * 16, k0), ar);
      axc = wmma_bf16(ax, lds_load_b(WT, 128, 256 + wave * 16, k0), axc);
      ahc = wmma_bf16(ah, lds_load_b(UT, 128, 256 + wave * 16, k0), ahc);
    }
    __syncthreads();

#pragma unroll
    for (int p = 0; p < 8; ++p) {
      int m = p + moff;
      float at = ab[m];
      float u  = at * sigm(az[p] + bz);
      float r  = sigm(ar[p] + br);
      float c  = tanh_f(axc[p] + bxc + r * (ahc[p] + bhc));
      float hn = (1.0f - u) * h[p] + u * c;
      h[p] = hn;
      hb[m * 128 + cz] = (bf16_t)hn;
    }
    wait_async0();
    __syncthreads();
  }

#pragma unroll
  for (int p = 0; p < 8; ++p)
    out[(size_t)(b0 + p + moff) * HH + cz] = h[p];
}

// ---------------------------------------------------------------------------
// Launch
// ---------------------------------------------------------------------------
extern "C" void kernel_launch(void* const* d_in, const int* in_sizes, int n_in,
                              void* d_out, int out_size, void* d_ws, size_t ws_size,
                              hipStream_t stream) {
  (void)in_sizes; (void)n_in; (void)out_size; (void)ws_size;

  const float* item   = (const float*)d_in[0];
  const float* xseq   = (const float*)d_in[1];
  const int*   mask   = (const int*)d_in[2];
  const float* gru_W  = (const float*)d_in[3];
  const float* gru_U  = (const float*)d_in[4];
  const float* gru_bx = (const float*)d_in[5];
  const float* gru_bh = (const float*)d_in[6];
  const float* att_Wi = (const float*)d_in[7];
  const float* att_bi = (const float*)d_in[8];
  const float* att_al = (const float*)d_in[9];
  const float* att_W1 = (const float*)d_in[10];
  const float* att_b1 = (const float*)d_in[11];
  const float* att_W2 = (const float*)d_in[12];
  const float* att_b2 = (const float*)d_in[13];
  const float* att_W3 = (const float*)d_in[14];
  const float* att_b3 = (const float*)d_in[15];
  const float* aug_W  = (const float*)d_in[16];
  const float* aug_U  = (const float*)d_in[17];
  const float* aug_bx = (const float*)d_in[18];
  const float* aug_bh = (const float*)d_in[19];

  float* out_final = (float*)d_out;                 // [B,H]
  float* out_feats = out_final + (size_t)BB * HH;   // [B,T,H]

  // Workspace: fp32 activations then bf16 transposed weights.
  float* q_ws      = (float*)d_ws;                  // [B,H]
  float* scores_ws = q_ws + (size_t)BB * HH;        // [B*T]
  float* att_ws    = scores_ws + (size_t)BB * TT;   // [B*T]
  bf16_t* wsb      = (bf16_t*)(att_ws + (size_t)BB * TT);
  bf16_t* gruWTg = wsb;                 // [384*128]
  bf16_t* gruUTg = gruWTg + 49152;      // [384*128]
  bf16_t* augWTg = gruUTg + 49152;      // [384*128]
  bf16_t* augUTg = augWTg + 49152;      // [384*128]
  bf16_t* WiTg   = augUTg + 49152;      // [128*128]
  bf16_t* W1Tg   = WiTg + 16384;        // [80*512]
  bf16_t* W2Tg   = W1Tg + 40960;        // [48*96] padded

  const size_t lds_rnn = (size_t)(2 * 384 * 128 + 2 * 16 * 128) * sizeof(bf16_t)
                       + 16 * sizeof(float) + 2 * 2048 * sizeof(float);
  const size_t lds_q   = (size_t)(128 * 128 + 16 * 128) * sizeof(bf16_t);
  const size_t lds_mlp = (size_t)(80 * 512 + 48 * 96 + 128 * 512 + 128 * 96) * sizeof(bf16_t)
                       + 48 * sizeof(float);

  // 0) One-time weight transpose + bf16 convert into workspace.
  conv_T_kernel<<<192, 256, 0, stream>>>(gru_W, gruWTg, 128, 384);
  conv_T_kernel<<<192, 256, 0, stream>>>(gru_U, gruUTg, 128, 384);
  conv_T_kernel<<<192, 256, 0, stream>>>(aug_W, augWTg, 128, 384);
  conv_T_kernel<<<192, 256, 0, stream>>>(aug_U, augUTg, 128, 384);
  conv_T_kernel<<<64, 256, 0, stream>>>(att_Wi, WiTg, 128, 128);
  conv_T_kernel<<<160, 256, 0, stream>>>(att_W1, W1Tg, 512, 80);
  conv_T_pad_kernel<<<18, 256, 0, stream>>>(att_W2, W2Tg, 80, 40, 96, 48);

  // 1) GRU -> feats (second output)
  gru_kernel<<<BB / 16, 256, lds_rnn, stream>>>(xseq, gruWTg, gruUTg, gru_bx, gru_bh, out_feats);
  // 2) q = PReLU(item @ Wi + bi)
  qproj_kernel<<<BB / 16, 256, lds_q, stream>>>(item, WiTg, att_bi, att_al, q_ws);
  // 3) attention MLP -> raw scores
  attmlp_kernel<<<(BB * TT) / 128, 256, lds_mlp, stream>>>(q_ws, out_feats,
                                                           W1Tg, att_b1, W2Tg, att_b2,
                                                           att_W3, att_b3, scores_ws);
  // 4) masked softmax over T
  softmax_kernel<<<BB, 256, 0, stream>>>(scores_ws, mask, att_ws);
  // 5) AUGRU -> final hidden state (first output)
  augru_kernel<<<BB / 16, 256, lds_rnn, stream>>>(out_feats, att_ws, aug_W ? augWTg : augWTg,
                                                  augUTg, aug_bx, aug_bh, out_final);
}
